// CausalSelfAttention_79688823210049
// MI455X (gfx1250) — matrix operand order repaired and checked by reading, not compiled, not run on hardware
//
#include <hip/hip_runtime.h>
#include <hip/hip_bf16.h>

// ---------------------------------------------------------------------------
// Causal self-attention for MI455X (gfx1250, wave32, WMMA, async-LDS staging).
// B=4, S=2048, E=1024, H=16, D=64.
// Pipeline: qkv = x@w_att (bf16 WMMA, f32 acc) -> flash attention (bf16 WMMA,
//           double-buffered global_load_async_to_lds staging) -> y@w_proj.
// ---------------------------------------------------------------------------

#define BDIM 4
#define SDIM 2048
#define EDIM 1024
#define HDIM 16
#define DDIM 64

typedef __attribute__((ext_vector_type(16))) __bf16 v16bf;
typedef __attribute__((ext_vector_type(8)))  float  v8f;

union Frag { v16bf v; unsigned int u[8]; __bf16 h[16]; };
union Pack4 { __bf16 h[4]; uint2 u2; };

static __device__ __forceinline__ v8f zero8() {
  v8f z = {0.f, 0.f, 0.f, 0.f, 0.f, 0.f, 0.f, 0.f};
  return z;
}

static __device__ __forceinline__ v8f wmma_bf16(v16bf a, v16bf b, v8f c) {
  // v_wmma_f32_16x16x32_bf16  (neg_a, A, neg_b, B, c_mod, C, reuse_a, reuse_b)
  return __builtin_amdgcn_wmma_f32_16x16x32_bf16(false, a, false, b, (short)0, c,
                                                 false, false);
}

// A-matrix 16x32 bf16 fragment: lane row = lane&15;
// lanes 0-15 hold K 0..15, lanes 16-31 hold K 16..31 (contiguous).
static __device__ __forceinline__ v16bf load_a_frag(const __bf16* base, int stride,
                                                    int lane) {
  const __bf16* row = base + (lane & 15) * stride + ((lane & 16) ? 16 : 0);
  Frag f;
  const unsigned int* p0 = (const unsigned int*)row;         // K kb+0..15
#pragma unroll
  for (int i = 0; i < 8; ++i) f.u[i] = p0[i];
  return f.v;
}

// B-matrix 32x16 bf16 fragment from [n][k] storage (contiguous K per lane):
// lane col = lane&15; lanes 0-15 hold K 0..15, lanes 16-31 hold K 16..31.
static __device__ __forceinline__ v16bf load_b_frag(const __bf16* base, int stride,
                                                    int lane) {
  const __bf16* row = base + (lane & 15) * stride + ((lane & 16) ? 16 : 0);
  Frag f;
  const unsigned int* p = (const unsigned int*)row;
#pragma unroll
  for (int i = 0; i < 8; ++i) f.u[i] = p[i];
  return f.v;
}

// B-matrix 32x16 bf16 fragment from [k][n]-major storage (K strided per lane):
// base must point at the (k-chunk, n-subtile) origin.
static __device__ __forceinline__ v16bf load_b_frag_strided(const __bf16* base,
                                                            int stride, int lane) {
  const __bf16* p = base + ((lane & 16) ? 16 : 0) * stride + (lane & 15);
  Frag f;
#pragma unroll
  for (int e = 0; e < 16; ++e) f.h[e] = p[e * stride];
  return f.v;
}

// ---------------------------------------------------------------------------
// Kernel 1: qkv = x @ w_att, scatter into Q/K/V [b,h,s,d] bf16; Q pre-scaled
// by 1/sqrt(D)=0.125. Block tile 128x128, K-step 32, 8 waves of 64x32 each.
// ---------------------------------------------------------------------------
#define AST 36
#define BST 36

__global__ __launch_bounds__(256) void qkv_gemm_kernel(
    const float* __restrict__ X, const float* __restrict__ W,
    __bf16* __restrict__ Q, __bf16* __restrict__ K, __bf16* __restrict__ V) {
  __shared__ __align__(16) __bf16 As[128 * AST];
  __shared__ __align__(16) __bf16 Bt[128 * BST];
  const int tid = threadIdx.x;
  const int lane = tid & 31;
  const int wid = tid >> 5;
  const int wm = wid & 1, wn = wid >> 1;
  const int m0 = blockIdx.y * 128;
  const int n0 = blockIdx.x * 128;
  const int hi8 = (lane & 16) ? 8 : 0;
  const int NW = 3 * EDIM;

  v8f acc[4][2];
#pragma unroll
  for (int i = 0; i < 4; ++i)
#pragma unroll
    for (int j = 0; j < 2; ++j) acc[i][j] = zero8();

  for (int k0 = 0; k0 < EDIM; k0 += 32) {
    // Prefetch next K-step tiles into near caches (WGP scope) while staging.
    if (k0 + 32 < EDIM) {
      __builtin_prefetch(X + (size_t)(m0 + (tid >> 1)) * EDIM + (k0 + 32) +
                             (tid & 1) * 16, 0, 3);
      __builtin_prefetch(W + (size_t)(k0 + 32 + (tid >> 3)) * NW + n0 +
                             (tid & 7) * 16, 0, 3);
    }
    // Stage A tile 128x32 (fp32 -> bf16), 4 contiguous floats per thread/iter.
#pragma unroll
    for (int it = 0; it < 4; ++it) {
      int i = tid + it * 256;               // 1024 quads
      int row = i >> 3, k = (i & 7) * 4;
      const float4 f = *(const float4*)(X + (size_t)(m0 + row) * EDIM + k0 + k);
      Pack4 pk;
      pk.h[0] = (__bf16)f.x; pk.h[1] = (__bf16)f.y;
      pk.h[2] = (__bf16)f.z; pk.h[3] = (__bf16)f.w;
      *(uint2*)&As[row * AST + k] = pk.u2;
    }
    // Stage B tile 32x128 transposed -> Bt[n][k] (fp32 -> bf16).
#pragma unroll
    for (int it = 0; it < 4; ++it) {
      int i = tid + it * 256;
      int k = i >> 5, n = (i & 31) * 4;
      const float4 f = *(const float4*)(W + (size_t)(k0 + k) * NW + n0 + n);
      Bt[(n + 0) * BST + k] = (__bf16)f.x;
      Bt[(n + 1) * BST + k] = (__bf16)f.y;
      Bt[(n + 2) * BST + k] = (__bf16)f.z;
      Bt[(n + 3) * BST + k] = (__bf16)f.w;
    }
    __syncthreads();

    v16bf af[4], bfr[2];
#pragma unroll
    for (int im = 0; im < 4; ++im)
      af[im] = load_a_frag(As + (wm * 64 + im * 16) * AST, AST, lane);
#pragma unroll
    for (int in = 0; in < 2; ++in)
      bfr[in] = load_b_frag(Bt + (wn * 32 + in * 16) * BST, BST, lane);
#pragma unroll
    for (int im = 0; im < 4; ++im)
#pragma unroll
      for (int in = 0; in < 2; ++in)
        acc[im][in] = wmma_bf16(af[im], bfr[in], acc[im][in]);
    __syncthreads();
  }

  // Scatter into Q/K/V [b,h,s,d] bf16.
#pragma unroll
  for (int im = 0; im < 4; ++im) {
#pragma unroll
    for (int in = 0; in < 2; ++in) {
      int j = n0 + wn * 32 + in * 16 + (lane & 15);
      int part = j >> 10, jj = j & 1023;
      int h = jj >> 6, d = jj & 63;
      __bf16* dst = (part == 0) ? Q : (part == 1) ? K : V;
      float scale = (part == 0) ? 0.125f : 1.0f;
#pragma unroll
      for (int r = 0; r < 8; ++r) {
        int m = m0 + wm * 64 + im * 16 + r + hi8;
        int bb = m >> 11, s = m & 2047;
        dst[((size_t)(bb * HDIM + h) * SDIM + s) * DDIM + d] =
            (__bf16)(acc[im][in][r] * scale);
      }
    }
  }
}

// ---------------------------------------------------------------------------
// Kernel 2: flash attention. Block = (b,h) x 128 q rows; wave w owns rows
// q0+16w..+15. K and V tiles (64x64 bf16, [key][d]) are staged with
// double-buffered GLOBAL_LOAD_ASYNC_TO_LDS_B128 (ASYNCcnt) so the next
// key-block's staging overlaps the current block's WMMA + softmax.
// ---------------------------------------------------------------------------
#define KST 72
#define PST 68
#define NEGBIG (-1.0e30f)

// Issue one 64x64 bf16 K tile + one V tile as async global->LDS copies.
// 512 16-byte granules per tile; 2 per thread per tile => 4 async ops/thread.
static __device__ __forceinline__ void issue_stage_async(
    const __bf16* Kg, const __bf16* Vg, __bf16* ksb, __bf16* vsb, int tid) {
#pragma unroll
  for (int it = 0; it < 2; ++it) {
    int i = tid + it * 256;                // 0..511
    int row = i >> 3, d = (i & 7) * 8;     // 64 rows x 8 granules
    unsigned kl = (unsigned)(size_t)&ksb[row * KST + d];   // LDS byte offset
    unsigned vl = (unsigned)(size_t)&vsb[row * KST + d];
    const __bf16* kg = Kg + row * DDIM + d;
    const __bf16* vg = Vg + row * DDIM + d;
    asm volatile("global_load_async_to_lds_b128 %0, %1, off"
                 :: "v"(kl), "v"(kg) : "memory");
    asm volatile("global_load_async_to_lds_b128 %0, %1, off"
                 :: "v"(vl), "v"(vg) : "memory");
  }
}

__global__ __launch_bounds__(256) void attn_kernel(
    const __bf16* __restrict__ Q, const __bf16* __restrict__ K,
    const __bf16* __restrict__ V, __bf16* __restrict__ Y) {
  __shared__ __align__(16) __bf16 Ks[2][64 * KST];
  __shared__ __align__(16) __bf16 Vs[2][64 * KST];
  __shared__ __align__(16) __bf16 Ps[8][16 * PST];
  const int tid = threadIdx.x;
  const int lane = tid & 31;
  const int w = tid >> 5;
  const int bh = blockIdx.y;               // b*H + h
  const int q0 = blockIdx.x * 128;
  const int qw0 = q0 + w * 16;             // wave's first q row
  const int hi8 = (lane & 16) ? 8 : 0;
  const __bf16* Kbase = K + (size_t)bh * SDIM * DDIM;
  const __bf16* Vbase = V + (size_t)bh * SDIM * DDIM;

  // Q fragments (already scaled by 1/8), held for the whole key loop.
  v16bf qf[2];
#pragma unroll
  for (int c = 0; c < 2; ++c)
    qf[c] = load_a_frag(Q + ((size_t)bh * SDIM + qw0) * DDIM + c * 32, DDIM, lane);

  v8f o[4];
#pragma unroll
  for (int i = 0; i < 4; ++i) o[i] = zero8();
  float mrow[8], lrow[8];
#pragma unroll
  for (int r = 0; r < 8; ++r) { mrow[r] = NEGBIG; lrow[r] = 0.f; }

  const int nkb = q0 / 64 + 2;             // uniform across block
  issue_stage_async(Kbase, Vbase, &Ks[0][0], &Vs[0][0], tid);   // block 0

  for (int kb = 0; kb < nkb; ++kb) {
    const int key0 = kb * 64;
    const int buf = kb & 1;
    if (kb + 1 < nkb) {
      // Start staging the next block into the other buffer, then wait for the
      // current block only (own async ops complete in order: <=4 outstanding
      // leaves just the 4 newly-issued ones in flight).
      issue_stage_async(Kbase + (size_t)(kb + 1) * 64 * DDIM,
                        Vbase + (size_t)(kb + 1) * 64 * DDIM,
                        &Ks[buf ^ 1][0], &Vs[buf ^ 1][0], tid);
      asm volatile("s_wait_asynccnt 0x4" ::: "memory");
    } else {
      asm volatile("s_wait_asynccnt 0x0" ::: "memory");
    }
    __syncthreads();

    // Scores: 16 q rows x 64 keys = 4 subtiles x 2 K-chunks of WMMA.
    v8f sc[4];
#pragma unroll
    for (int sub = 0; sub < 4; ++sub) {
      v8f s = zero8();
#pragma unroll
      for (int c = 0; c < 2; ++c) {
        v16bf kf = load_b_frag(&Ks[buf][0] + (sub * 16) * KST + c * 32, KST, lane);
        s = wmma_bf16(qf[c], kf, s);
      }
      int key = key0 + sub * 16 + (lane & 15);
#pragma unroll
      for (int r = 0; r < 8; ++r) {
        int q = qw0 + r + hi8;
        if (key > q) s[r] = NEGBIG;        // causal mask
      }
      sc[sub] = s;
    }

    // Row max: per-lane max across subtiles, then butterfly over the 16-lane
    // group holding the same rows (xor 8,4,2,1 stays within each half-wave).
    float bmax[8];
#pragma unroll
    for (int r = 0; r < 8; ++r)
      bmax[r] = fmaxf(fmaxf(sc[0][r], sc[1][r]), fmaxf(sc[2][r], sc[3][r]));
#pragma unroll
    for (int off = 8; off >= 1; off >>= 1)
#pragma unroll
      for (int r = 0; r < 8; ++r)
        bmax[r] = fmaxf(bmax[r], __shfl_xor(bmax[r], off, 32));

    float mnew[8], alpha[8];
#pragma unroll
    for (int r = 0; r < 8; ++r) {
      mnew[r] = fmaxf(mrow[r], bmax[r]);
      alpha[r] = __expf(mrow[r] - mnew[r]);
      mrow[r] = mnew[r];
    }

    // P = exp(s - mnew); accumulate row sums.
    float psum[8];
#pragma unroll
    for (int r = 0; r < 8; ++r) psum[r] = 0.f;
#pragma unroll
    for (int sub = 0; sub < 4; ++sub)
#pragma unroll
      for (int r = 0; r < 8; ++r) {
        float p = __expf(sc[sub][r] - mnew[r]);
        sc[sub][r] = p;
        psum[r] += p;
      }
#pragma unroll
    for (int off = 8; off >= 1; off >>= 1)
#pragma unroll
      for (int r = 0; r < 8; ++r)
        psum[r] += __shfl_xor(psum[r], off, 32);
#pragma unroll
    for (int r = 0; r < 8; ++r) lrow[r] = lrow[r] * alpha[r] + psum[r];

    // Rescale running output.
#pragma unroll
    for (int dsub = 0; dsub < 4; ++dsub)
#pragma unroll
      for (int r = 0; r < 8; ++r) o[dsub][r] *= alpha[r];

    // Store P (bf16) to this wave's LDS region in [row][key] layout
    // (C-layout -> A-layout conversion via LDS).
#pragma unroll
    for (int sub = 0; sub < 4; ++sub)
#pragma unroll
      for (int r = 0; r < 8; ++r)
        Ps[w][(r + hi8) * PST + sub * 16 + (lane & 15)] = (__bf16)sc[sub][r];
    __syncthreads();

    // O += P @ V. V stays in [key][d] layout; B-frags gathered K-strided.
#pragma unroll
    for (int c = 0; c < 2; ++c) {
      v16bf pa = load_a_frag(&Ps[w][0] + c * 32, PST, lane);
#pragma unroll
      for (int dsub = 0; dsub < 4; ++dsub) {
        v16bf vb = load_b_frag_strided(&Vs[buf][0] + (c * 32) * KST + dsub * 16,
                                       KST, lane);
        o[dsub] = wmma_bf16(pa, vb, o[dsub]);
      }
    }
    __syncthreads();   // all waves done with buf before it is overwritten
  }

  // Normalize and store y as bf16 in [b, s, e] (e = h*64 + d).
  const int bb = bh >> 4, h = bh & 15;
#pragma unroll
  for (int dsub = 0; dsub < 4; ++dsub) {
    int d = dsub * 16 + (lane & 15);
#pragma unroll
    for (int r = 0; r < 8; ++r) {
      int q = qw0 + r + hi8;
      float val = o[dsub][r] / lrow[r];
      Y[((size_t)(bb * SDIM + q) * HDIM + h) * DDIM + d] = (__bf16)val;
    }
  }
}

// ---------------------------------------------------------------------------
// Kernel 3: out = y @ w_proj (A bf16, B fp32->bf16, out fp32).
// ---------------------------------------------------------------------------
__global__ __launch_bounds__(256) void proj_gemm_kernel(
    const __bf16* __restrict__ Y, const float* __restrict__ W,
    float* __restrict__ O) {
  __shared__ __align__(16) __bf16 As[128 * AST];
  __shared__ __align__(16) __bf16 Bt[128 * BST];
  const int tid = threadIdx.x;
  const int lane = tid & 31;
  const int wid = tid >> 5;
  const int wm = wid & 1, wn = wid >> 1;
  const int m0 = blockIdx.y * 128;
  const int n0 = blockIdx.x * 128;
  const int hi8 = (lane & 16) ? 8 : 0;

  v8f acc[4][2];
#pragma unroll
  for (int i = 0; i < 4; ++i)
#pragma unroll
    for (int j = 0; j < 2; ++j) acc[i][j] = zero8();

  for (int k0 = 0; k0 < EDIM; k0 += 32) {
    if (k0 + 32 < EDIM) {
      __builtin_prefetch(Y + (size_t)(m0 + (tid >> 1)) * EDIM + (k0 + 32) +
                             (tid & 1) * 16, 0, 3);
      __builtin_prefetch(W + (size_t)(k0 + 32 + (tid >> 3)) * EDIM + n0 +
                             (tid & 7) * 16, 0, 3);
    }
#pragma unroll
    for (int it = 0; it < 4; ++it) {
      int i = tid + it * 256;
      int row = i >> 3, k = (i & 7) * 4;
      *(uint2*)&As[row * AST + k] =
          *(const uint2*)(Y + (size_t)(m0 + row) * EDIM + k0 + k);
    }
#pragma unroll
    for (int it = 0; it < 4; ++it) {
      int i = tid + it * 256;
      int k = i >> 5, n = (i & 31) * 4;
      const float4 f = *(const float4*)(W + (size_t)(k0 + k) * EDIM + n0 + n);
      Bt[(n + 0) * BST + k] = (__bf16)f.x;
      Bt[(n + 1) * BST + k] = (__bf16)f.y;
      Bt[(n + 2) * BST + k] = (__bf16)f.z;
      Bt[(n + 3) * BST + k] = (__bf16)f.w;
    }
    __syncthreads();

    v16bf af[4], bfr[2];
#pragma unroll
    for (int im = 0; im < 4; ++im)
      af[im] = load_a_frag(As + (wm * 64 + im * 16) * AST, AST, lane);
#pragma unroll
    for (int in = 0; in < 2; ++in)
      bfr[in] = load_b_frag(Bt + (wn * 32 + in * 16) * BST, BST, lane);
#pragma unroll
    for (int im = 0; im < 4; ++im)
#pragma unroll
      for (int in = 0; in < 2; ++in)
        acc[im][in] = wmma_bf16(af[im], bfr[in], acc[im][in]);
    __syncthreads();
  }

#pragma unroll
  for (int im = 0; im < 4; ++im) {
#pragma unroll
    for (int in = 0; in < 2; ++in) {
      int j = n0 + wn * 32 + in * 16 + (lane & 15);
#pragma unroll
      for (int r = 0; r < 8; ++r) {
        int m = m0 + wm * 64 + im * 16 + r + hi8;
        O[(size_t)m * EDIM + j] = acc[im][in][r];
      }
    }
  }
}

// ---------------------------------------------------------------------------
extern "C" void kernel_launch(void* const* d_in, const int* in_sizes, int n_in,
                              void* d_out, int out_size, void* d_ws, size_t ws_size,
                              hipStream_t stream) {
  (void)in_sizes; (void)n_in; (void)out_size; (void)ws_size;
  const float* x      = (const float*)d_in[0];
  const float* w_att  = (const float*)d_in[1];
  const float* w_proj = (const float*)d_in[2];
  float* out          = (float*)d_out;

  const size_t NQKV = (size_t)BDIM * HDIM * SDIM * DDIM;  // 8,388,608 elems
  __bf16* qw = (__bf16*)d_ws;
  __bf16* kw = qw + NQKV;
  __bf16* vw = kw + NQKV;
  __bf16* yw = vw + NQKV;                                 // 64 MB total bf16 ws

  // qkv projection: M=8192, N=3072
  qkv_gemm_kernel<<<dim3(3 * EDIM / 128, BDIM * SDIM / 128), 256, 0, stream>>>(
      x, w_att, qw, kw, vw);
  // flash attention: 16 q-tiles x 64 (b,h)
  attn_kernel<<<dim3(SDIM / 128, BDIM * HDIM), 256, 0, stream>>>(qw, kw, vw, yw);
  // output projection: M=8192, N=1024
  proj_gemm_kernel<<<dim3(EDIM / 128, BDIM * SDIM / 128), 256, 0, stream>>>(
      yw, w_proj, out);
}
